// BiMambaBlock_40939628266180
// MI455X (gfx1250) — compile-verified, matrix operand order checked by reading, not run on hardware
//
#include <hip/hip_runtime.h>
#include <hip/hip_bf16.h>

// ---------------- problem constants ----------------
#define BB      4
#define LL      2048
#define DMODEL  1024
#define DINNER  2048
#define NH      32
#define HD      64
#define DSTATE  64
#define DCONV   4
#define CONVD   2176          // D_INNER + 2*D_STATE
#define NPROJ   4256          // 2*D_INNER + 2*D_STATE + NHEADS
#define ZX_LD   4352          // NPROJ padded to a multiple of 128 for WMMA tiling
#define ROWS    (BB * LL)     // 8192

// ---------------- vector types ----------------
typedef __bf16        v16bf __attribute__((ext_vector_type(16)));
typedef float         v8f   __attribute__((ext_vector_type(8)));
typedef unsigned int  v8u   __attribute__((ext_vector_type(8)));
typedef unsigned int  v4u   __attribute__((ext_vector_type(4)));
typedef int           v4i   __attribute__((ext_vector_type(4)));
typedef int           v8i   __attribute__((ext_vector_type(8)));

__device__ __forceinline__ unsigned short f2bf(float f) {
    unsigned int u = __builtin_bit_cast(unsigned int, f);
    unsigned int r = (u + 0x7FFFu + ((u >> 16) & 1u)) >> 16;   // round-to-nearest-even
    return (unsigned short)r;
}

__device__ __forceinline__ float sigmoidf_(float x) { return 1.f / (1.f + __expf(-x)); }

// ---------------- fp32 -> bf16 convert (row-major, optional col padding) ----------------
__global__ __launch_bounds__(256) void cvt_bf16_pad(const float* __restrict__ src,
                                                    unsigned short* __restrict__ dst,
                                                    int total, int srcCols, int dstCols) {
    int i = blockIdx.x * blockDim.x + threadIdx.x;
    if (i >= total) return;
    int col = i % dstCols;
    int row = i / dstCols;
    float v = (col < srcCols) ? src[(size_t)row * srcCols + col] : 0.f;
    dst[i] = f2bf(v);
}

// ---------------- fp32 -> bf16 transpose: src(K x Ncols) -> dst(Npad x K) ----------------
// Pre-transposing the weights makes the GEMM's B tiles identical in shape to A tiles
// (row-major 128x32), so both can be staged by the Tensor Data Mover.
__global__ __launch_bounds__(256) void cvt_bf16_transpose(const float* __restrict__ src,
                                                          unsigned short* __restrict__ dst,
                                                          int K, int Ncols, int Npad) {
    int i = blockIdx.x * blockDim.x + threadIdx.x;
    int total = Npad * K;
    if (i >= total) return;
    int k = i % K;
    int n = i / K;
    float v = (n < Ncols) ? src[(size_t)k * Ncols + n] : 0.f;
    dst[i] = f2bf(v);
}

// ---------------- bf16 WMMA GEMM: C(MxN,f32) = A(MxK) * Bt(NxK)^T ----------------
#define BM  128
#define BN  128
#define BKS 32
#define LDT 34   // LDS row stride in ushorts (+1 DWORD pad per 64B row)

#if defined(__has_builtin)
# if __has_builtin(__builtin_amdgcn_tensor_load_to_lds)
#  define USE_TDM 1
# endif
#endif
#ifndef USE_TDM
# define USE_TDM 0
#endif

#if USE_TDM
// Generic LDS pointer -> LDS byte offset (ISA 10.2: LDS aperture addr = addr[31:0]).
__device__ __forceinline__ unsigned lds_offset(const void* p) {
    return (unsigned)(unsigned long long)p;
}

// Issue one TDM 2-D tile load: (BM rows) x (BKS bf16) from a row-major tensor with
// row stride `strideElems`, into LDS at `ldsOff` with +1 DWORD pad per 64B row
// (matches LDT=34). D# layout per CDNA5 ISA ch.8.
__device__ __forceinline__ void tdm_load_tile(const unsigned short* gtile,
                                              unsigned ldsOff,
                                              unsigned long long strideElems,
                                              unsigned long long nRowsTotal) {
    unsigned long long ga = (unsigned long long)gtile;
    v4u g0;
    g0[0] = 1u;                                   // count=1, user descriptor
    g0[1] = ldsOff;                               // lds_addr [63:32]
    g0[2] = (unsigned)ga;                         // global_addr[31:0]
    g0[3] = (unsigned)(ga >> 32) | (2u << 30);    // global_addr[56:32] | type=2
    union { unsigned long long q[4]; v8i v; } g1;
    g1.q[0] = g1.q[1] = g1.q[2] = g1.q[3] = 0;
    // dword0: data_size=1(2B)<<16 | pad_enable<<20 | pad_interval=3(16 DWORDs)<<22 | pad_amount=0(1 DWORD)<<25
    g1.q[0] |= (1ull << 16) | (1ull << 20) | (3ull << 22);
    g1.q[0] |= (strideElems & 0xFFFFull) << 48;           // tensor_dim0 [79:48] low 16
    g1.q[1] |= (strideElems >> 16) & 0xFFFFull;           //              high bits
    g1.q[1] |= (nRowsTotal & 0xFFFFFFFFull) << 16;        // tensor_dim1 [111:80]
    g1.q[1] |= (unsigned long long)BKS << 48;             // tile_dim0   [127:112]
    g1.q[2] |= (unsigned long long)BM;                    // tile_dim1   [143:128]
    g1.q[2] |= (strideElems & 0xFFFFFFFFull) << 32;       // dim0_stride [207:160] low 32
    g1.q[3] |= (strideElems >> 32) & 0xFFFFull;           //              high 16
    const v4i z4 = {0, 0, 0, 0};
#if __clang_major__ >= 23
    const v8i z8 = {0, 0, 0, 0, 0, 0, 0, 0};
    __builtin_amdgcn_tensor_load_to_lds(g0, g1.v, z4, z4, z8, 0);
#else
    __builtin_amdgcn_tensor_load_to_lds(g0, g1.v, z4, z4, 0);
#endif
}
#endif // USE_TDM

// One K-step of WMMA work for a wave: gather A/B fragments per the ISA 16-bit
// layout (lane = (r, hi); element pair j -> k = (j<4 ? 2j : 16+2(j-4)) + 8*hi)
// and run the 2x4 grid of v_wmma_f32_16x16x32_bf16.
__device__ __forceinline__ void wmma_tile_step(const unsigned short* pA,
                                               const unsigned short* pB,
                                               v8f acc[2][4],
                                               int waveM, int waveN, int r, int hi) {
    const int kb = hi * 8;
    v8u au[2], bu[4];
#pragma unroll
    for (int mt = 0; mt < 2; ++mt) {
        const unsigned short* rp = pA + (waveM * 32 + mt * 16 + r) * LDT;
#pragma unroll
        for (int j = 0; j < 4; ++j) {
            au[mt][j]     = *(const unsigned int*)(rp + kb + 2 * j);
            au[mt][j + 4] = *(const unsigned int*)(rp + 16 + kb + 2 * j);
        }
    }
#pragma unroll
    for (int nt = 0; nt < 4; ++nt) {
        const unsigned short* rp = pB + (waveN * 64 + nt * 16 + r) * LDT;
#pragma unroll
        for (int j = 0; j < 4; ++j) {
            bu[nt][j]     = *(const unsigned int*)(rp + kb + 2 * j);
            bu[nt][j + 4] = *(const unsigned int*)(rp + 16 + kb + 2 * j);
        }
    }
#pragma unroll
    for (int mt = 0; mt < 2; ++mt) {
        v16bf a = __builtin_bit_cast(v16bf, au[mt]);
#pragma unroll
        for (int nt = 0; nt < 4; ++nt) {
            v16bf b = __builtin_bit_cast(v16bf, bu[nt]);
            acc[mt][nt] = __builtin_amdgcn_wmma_f32_16x16x32_bf16(
                false, a, false, b, (short)0, acc[mt][nt], false, false);
        }
    }
}

__global__ __launch_bounds__(256) void gemm_bf16_wmma(const unsigned short* __restrict__ A,
                                                      const unsigned short* __restrict__ Bt,
                                                      float* __restrict__ C,
                                                      int M, int N, int K) {
    const int tid   = threadIdx.x;
    const int lane  = tid & 31;
    const int wave  = tid >> 5;
    const int waveM = wave & 3;
    const int waveN = wave >> 2;
    const int r     = lane & 15;
    const int hi    = lane >> 4;

    const int m0 = blockIdx.y * BM;
    const int n0 = blockIdx.x * BN;

    v8f acc[2][4];
    const v8f vzero = {0.f, 0.f, 0.f, 0.f, 0.f, 0.f, 0.f, 0.f};
#pragma unroll
    for (int mt = 0; mt < 2; ++mt)
#pragma unroll
        for (int nt = 0; nt < 4; ++nt) acc[mt][nt] = vzero;

#if USE_TDM
    // Double-buffered tiles, filled by the Tensor Data Mover (TENSORcnt-tracked),
    // so the whole K-loop fetch pipeline runs concurrently with the WMMAs.
    __shared__ unsigned short sA[2][BM * LDT];
    __shared__ unsigned short sB[2][BN * LDT];

    const unsigned ldsA[2] = { lds_offset(&sA[0][0]), lds_offset(&sA[1][0]) };
    const unsigned ldsB[2] = { lds_offset(&sB[0][0]), lds_offset(&sB[1][0]) };
    const bool tdmWave = (tid < 32);
    const int nk = K / BKS;

    if (tdmWave) {
        tdm_load_tile(A  + (size_t)m0 * K, ldsA[0], (unsigned long long)K, (unsigned long long)M);
        tdm_load_tile(Bt + (size_t)n0 * K, ldsB[0], (unsigned long long)K, (unsigned long long)N);
    }
    for (int i = 0; i < nk; ++i) {
        if (tdmWave) {
            if (i + 1 < nk) {
                const int kn = (i + 1) * BKS;
                tdm_load_tile(A  + (size_t)m0 * K + kn, ldsA[(i + 1) & 1],
                              (unsigned long long)K, (unsigned long long)M);
                tdm_load_tile(Bt + (size_t)n0 * K + kn, ldsB[(i + 1) & 1],
                              (unsigned long long)K, (unsigned long long)N);
                __builtin_amdgcn_s_wait_tensorcnt(2);   // tile i landed; tile i+1 in flight
            } else {
                __builtin_amdgcn_s_wait_tensorcnt(0);
            }
        }
        __syncthreads();
        wmma_tile_step(&sA[i & 1][0], &sB[i & 1][0], acc, waveM, waveN, r, hi);
        __syncthreads();
    }
#else
    // Manual staging fallback: identical row-major 128x32 tiles for A and Bt.
    __shared__ unsigned short sA[BM * LDT];
    __shared__ unsigned short sB[BN * LDT];

    for (int kt = 0; kt < K; kt += BKS) {
#pragma unroll
        for (int s = 0; s < 2; ++s) {
            int q = tid + s * 256;
            int row = q >> 2;
            int kc = (q & 3) * 8;
            {
                const uint4 d = *(const uint4*)(A + (size_t)(m0 + row) * K + kt + kc);
                unsigned int* dst = (unsigned int*)&sA[row * LDT + kc];
                dst[0] = d.x; dst[1] = d.y; dst[2] = d.z; dst[3] = d.w;
            }
            {
                const uint4 d = *(const uint4*)(Bt + (size_t)(n0 + row) * K + kt + kc);
                unsigned int* dst = (unsigned int*)&sB[row * LDT + kc];
                dst[0] = d.x; dst[1] = d.y; dst[2] = d.z; dst[3] = d.w;
            }
        }
        __syncthreads();
        if (kt + BKS < K) {
            int row = tid >> 2, kc = (tid & 3) * 8;
            __builtin_prefetch(A  + (size_t)(m0 + row) * K + kt + BKS + kc, 0, 0);
            __builtin_prefetch(Bt + (size_t)(n0 + row) * K + kt + BKS + kc, 0, 0);
        }
        wmma_tile_step(&sA[0], &sB[0], acc, waveM, waveN, r, hi);
        __syncthreads();
    }
#endif

    // Epilogue: C/D layout -> lane holds (row = j + 8*hi, col = r).
#pragma unroll
    for (int mt = 0; mt < 2; ++mt)
#pragma unroll
        for (int nt = 0; nt < 4; ++nt) {
            const int col = n0 + waveN * 64 + nt * 16 + r;
#pragma unroll
            for (int j = 0; j < 8; ++j) {
                const int row = m0 + waveM * 32 + mt * 16 + j + 8 * hi;
                C[(size_t)row * N + col] = acc[mt][nt][j];
            }
        }
}

// ---------------- depthwise causal conv (k=4) + SiLU ----------------
__global__ __launch_bounds__(256) void conv_silu_kernel(const float* __restrict__ zx,
                                                        const float* __restrict__ cw,
                                                        const float* __restrict__ cb,
                                                        float* __restrict__ xc) {
    int idx = blockIdx.x * blockDim.x + threadIdx.x;
    const int total = BB * LL * CONVD;
    if (idx >= total) return;
    int c = idx % CONVD;
    int l = (idx / CONVD) % LL;
    int b = idx / (CONVD * LL);
    const float* src = zx + (size_t)(b * LL) * ZX_LD + DINNER + c;  // xBC slice
    float acc = cb[c];
#pragma unroll
    for (int k = 0; k < DCONV; ++k) {
        int ll = l - (DCONV - 1) + k;
        if (ll >= 0) acc += cw[k * CONVD + c] * src[(size_t)ll * ZX_LD];
    }
    xc[idx] = acc * sigmoidf_(acc);
}

// ---------------- sequential SSM scan: one block per (batch, head) ----------------
__global__ __launch_bounds__(64) void scan_kernel(const float* __restrict__ zx,
                                                  const float* __restrict__ xc,
                                                  const float* __restrict__ dt_bias,
                                                  const float* __restrict__ A_log,
                                                  const float* __restrict__ Dvec,
                                                  float* __restrict__ y) {
    const int bh = blockIdx.x;
    const int b  = bh >> 5;
    const int h  = bh & 31;
    const int p  = threadIdx.x;

    __shared__ float sB[DSTATE];
    __shared__ float sC[DSTATE];
    __shared__ float sdt;

    const float Ah  = -__expf(A_log[h]);
    const float dtb = dt_bias[h];
    const float Dh  = Dvec[h];

    float hs[DSTATE];
#pragma unroll
    for (int n = 0; n < DSTATE; ++n) hs[n] = 0.f;

    for (int t = 0; t < LL; ++t) {
        const size_t rowc = (size_t)(b * LL + t) * CONVD;
        sB[p] = xc[rowc + DINNER + p];
        sC[p] = xc[rowc + DINNER + DSTATE + p];
        if (p == 0) {
            float raw = zx[(size_t)(b * LL + t) * ZX_LD + (DINNER + CONVD) + h] + dtb;
            sdt = (raw > 20.f) ? raw : log1pf(__expf(raw));   // softplus
        }
        const float xp = xc[rowc + h * HD + p];
        __syncthreads();

        const float dt = sdt;
        const float dA = __expf(dt * Ah);
        const float cf = dt * xp;
        float yp = 0.f;
#pragma unroll
        for (int n = 0; n < DSTATE; ++n) {
            hs[n] = hs[n] * dA + cf * sB[n];
            yp += hs[n] * sC[n];
        }
        y[(size_t)(b * LL + t) * DINNER + h * HD + p] = yp + xp * Dh;
        __syncthreads();
    }
}

// ---------------- gate (SiLU(z)) + RMSNorm, emit bf16 for GEMM2 ----------------
__global__ __launch_bounds__(256) void gate_norm_kernel(const float* __restrict__ y,
                                                        const float* __restrict__ zx,
                                                        const float* __restrict__ norm_w,
                                                        unsigned short* __restrict__ ybf) {
    const int row = blockIdx.x;
    const int tid = threadIdx.x;
    __shared__ float red[256];
    const float* yr = y + (size_t)row * DINNER;
    const float* zr = zx + (size_t)row * ZX_LD;

    float vals[8];
    float ss = 0.f;
#pragma unroll
    for (int j = 0; j < 8; ++j) {
        const int c = tid + j * 256;
        const float z = zr[c];
        const float v = yr[c] * (z * sigmoidf_(z));
        vals[j] = v;
        ss += v * v;
    }
    red[tid] = ss;
    __syncthreads();
    for (int s = 128; s > 0; s >>= 1) {
        if (tid < s) red[tid] += red[tid + s];
        __syncthreads();
    }
    const float scale = rsqrtf(red[0] / (float)DINNER + 1e-5f);
#pragma unroll
    for (int j = 0; j < 8; ++j) {
        const int c = tid + j * 256;
        ybf[(size_t)row * DINNER + c] = f2bf(vals[j] * scale * norm_w[c]);
    }
}

// ---------------- launch ----------------
extern "C" void kernel_launch(void* const* d_in, const int* in_sizes, int n_in,
                              void* d_out, int out_size, void* d_ws, size_t ws_size,
                              hipStream_t stream) {
    const float* u       = (const float*)d_in[0];
    const float* W_in    = (const float*)d_in[1];
    const float* conv_w  = (const float*)d_in[2];
    const float* conv_b  = (const float*)d_in[3];
    const float* dt_bias = (const float*)d_in[4];
    const float* A_log   = (const float*)d_in[5];
    const float* Dv      = (const float*)d_in[6];
    const float* norm_w  = (const float*)d_in[7];
    const float* W_out   = (const float*)d_in[8];

    char* ws = (char*)d_ws;
    size_t off = 0;
    auto carve = [&](size_t bytes) {
        void* p = ws + off;
        off = (off + bytes + 255) & ~(size_t)255;
        return p;
    };
    unsigned short* u_bf   = (unsigned short*)carve((size_t)ROWS * DMODEL * 2);
    unsigned short* win_t  = (unsigned short*)carve((size_t)ZX_LD * DMODEL * 2);  // 4352 x 1024 (W_in^T)
    unsigned short* wout_t = (unsigned short*)carve((size_t)DMODEL * DINNER * 2); // 1024 x 2048 (W_out^T)
    float*          zx     = (float*)carve((size_t)ROWS * ZX_LD * 4);
    float*          xc     = (float*)carve((size_t)ROWS * CONVD * 4);
    float*          ybuf   = (float*)carve((size_t)ROWS * DINNER * 4);
    unsigned short* ybf    = (unsigned short*)carve((size_t)ROWS * DINNER * 2);

    // 1) precision conversion: activations row-major bf16, weights transposed bf16
    {
        int t0 = ROWS * DMODEL;
        cvt_bf16_pad<<<(t0 + 255) / 256, 256, 0, stream>>>(u, u_bf, t0, DMODEL, DMODEL);
        int t1 = ZX_LD * DMODEL;
        cvt_bf16_transpose<<<(t1 + 255) / 256, 256, 0, stream>>>(W_in, win_t, DMODEL, NPROJ, ZX_LD);
        int t2 = DMODEL * DINNER;
        cvt_bf16_transpose<<<(t2 + 255) / 256, 256, 0, stream>>>(W_out, wout_t, DINNER, DMODEL, DMODEL);
    }

    // 2) zxbcdt = u @ W_in   (8192 x 4352 x 1024, TDM-staged bf16 WMMA, f32 accumulate)
    gemm_bf16_wmma<<<dim3(ZX_LD / BN, ROWS / BM), 256, 0, stream>>>(
        u_bf, win_t, zx, ROWS, ZX_LD, DMODEL);

    // 3) depthwise causal conv + SiLU on xBC slice
    {
        int total = BB * LL * CONVD;
        conv_silu_kernel<<<(total + 255) / 256, 256, 0, stream>>>(zx, conv_w, conv_b, xc);
    }

    // 4) sequential selective-state scan (128 blocks = B*NHEADS, 64 lanes each)
    scan_kernel<<<BB * NH, 64, 0, stream>>>(zx, xc, dt_bias, A_log, Dv, ybuf);

    // 5) gate + RMSNorm -> bf16
    gate_norm_kernel<<<ROWS, 256, 0, stream>>>(ybuf, zx, norm_w, ybf);

    // 6) out = y @ W_out  (8192 x 1024 x 2048, f32 out to d_out)
    gemm_bf16_wmma<<<dim3(DMODEL / BN, ROWS / BM), 256, 0, stream>>>(
        ybf, wout_t, (float*)d_out, ROWS, DMODEL, DINNER);
}
